// Generator_5291399709210
// MI455X (gfx1250) — matrix operand order repaired and checked by reading, not compiled, not run on hardware
//
#include <hip/hip_runtime.h>
#include <math.h>

// ---------------------------------------------------------------------------
// Types for CDNA5 WMMA
// ---------------------------------------------------------------------------
typedef __attribute__((ext_vector_type(16))) _Float16 v16h;
typedef __attribute__((ext_vector_type(8)))  float    v8f;
typedef __attribute__((ext_vector_type(4)))  float    vf4;

#define GAIN 1.41421356237309515f
#define INV_SQRT512 0.044194173824159216f

// ---------------------------------------------------------------------------
// Deterministic hash-based Gaussian noise (stand-in for jax threefry normal)
// ---------------------------------------------------------------------------
__device__ __forceinline__ unsigned hash_u32(unsigned x) {
    x ^= x >> 16; x *= 0x7feb352dU;
    x ^= x >> 15; x *= 0x846ca68bU;
    x ^= x >> 16;
    return x;
}

__device__ __forceinline__ float gaussian_np(unsigned seed, unsigned b, unsigned p) {
    unsigned h1 = hash_u32(seed * 0x9e3779b9U ^ b * 0x85ebca6bU ^ p * 0xc2b2ae35U ^ 0x27d4eb2fU);
    unsigned h2 = hash_u32(h1 ^ 0x165667b1U);
    float u1 = (float)(h1 >> 8) * (1.0f / 16777216.0f);
    float u2 = (float)(h2 >> 8) * (1.0f / 16777216.0f);
    u1 = fmaxf(u1, 1e-7f);
    return sqrtf(-2.0f * __logf(u1)) * __cosf(6.28318530718f * u2);
}

// ---------------------------------------------------------------------------
// pixnorm: out[b,:] = z[b,:] * rsqrt(mean(z^2) + 1e-8); one block per batch row
// ---------------------------------------------------------------------------
__global__ void pixnorm_kernel(const float* __restrict__ z, float* __restrict__ out) {
    __shared__ float red[256];
    const int b = blockIdx.x;
    const int t = threadIdx.x;
    float v0 = z[b * 512 + t];
    float v1 = z[b * 512 + 256 + t];
    red[t] = v0 * v0 + v1 * v1;
    __syncthreads();
    for (int s = 128; s > 0; s >>= 1) {
        if (t < s) red[t] += red[t + s];
        __syncthreads();
    }
    float r = rsqrtf(red[0] / 512.0f + 1e-8f);
    out[b * 512 + t]       = v0 * r;
    out[b * 512 + 256 + t] = v1 * r;
}

// ---------------------------------------------------------------------------
// Equalized linear: out[b,n] = x[b,:] . w[n,:] * scale + bias[n]  (opt lrelu*g)
// Used for the mapping net (doAct=1) and style affines (doAct=0). K=512 always.
// ---------------------------------------------------------------------------
__global__ void eqlin_kernel(const float* __restrict__ x, const float* __restrict__ w,
                             const float* __restrict__ bias, float* __restrict__ out,
                             int N, int K, float scale, int doAct) {
    int t = blockIdx.x * blockDim.x + threadIdx.x;
    if (t >= 8 * N) return;
    int b = t / N, n = t - b * N;
    const vf4* xr = (const vf4*)(x + (size_t)b * K);
    const vf4* wr = (const vf4*)(w + (size_t)n * K);
    float acc = 0.0f;
    for (int k = 0; k < K / 4; ++k) {
        vf4 a = xr[k], c = wr[k];
        acc += a.x * c.x + a.y * c.y + a.z * c.z + a.w * c.w;
    }
    float v = acc * scale + bias[n];
    if (doAct) v = (v > 0.0f ? v : 0.2f * v) * GAIN;
    out[(size_t)b * N + n] = v;
}

// ---------------------------------------------------------------------------
// Weight prep: f32 (Cout,Cin,k,k) -> f16 GEMM matrix (Cout x Cin*k2) and
// per (co,ci) sum of squares for the demod reduction.
// ---------------------------------------------------------------------------
__global__ void wprep_kernel(const float* __restrict__ w, _Float16* __restrict__ w16,
                             float* __restrict__ wsq, int Cout, int Cin, int K2) {
    int t = blockIdx.x * blockDim.x + threadIdx.x;
    if (t >= Cout * Cin) return;
    const float* src = w + (size_t)t * K2;
    _Float16*    dst = w16 + (size_t)t * K2;
    float s = 0.0f;
    for (int q = 0; q < K2; ++q) {
        float v = src[q];
        s += v * v;
        dst[q] = (_Float16)v;
    }
    wsq[t] = s;
}

// ---------------------------------------------------------------------------
// Demod: d[b,co] = rsqrt( wscale^2 * sum_ci wsq[co,ci]*s[b,ci]^2 + 1e-8 )
// ---------------------------------------------------------------------------
__global__ void demod_kernel(const float* __restrict__ s, const float* __restrict__ wsq,
                             float* __restrict__ d, int Cout, int Cin, float wscale) {
    int t = blockIdx.x * blockDim.x + threadIdx.x;
    if (t >= 8 * Cout) return;
    int b = t / Cout, co = t - b * Cout;
    const float* wr = wsq + (size_t)co * Cin;
    const float* sr = s + (size_t)b * Cin;
    float acc = 0.0f;
    for (int ci = 0; ci < Cin; ++ci) {
        float sv = sr[ci];
        acc += wr[ci] * sv * sv;
    }
    d[t] = rsqrtf(wscale * wscale * acc + 1e-8f);
}

// ---------------------------------------------------------------------------
// Block-0 init: h[b,c,p] = const[c,p] + N0(b,p)*noise1[c]   (4x4, 512 ch)
// ---------------------------------------------------------------------------
__global__ void const_noise_kernel(const float* __restrict__ cst,
                                   const float* __restrict__ ns,
                                   float* __restrict__ h) {
    int t = blockIdx.x * blockDim.x + threadIdx.x;
    if (t >= 8 * 512 * 16) return;
    int p = t & 15;
    int c = (t >> 4) & 511;
    int b = t >> 13;
    h[t] = cst[c * 16 + p] + gaussian_np(0u, (unsigned)b, (unsigned)p) * ns[c];
}

// ---------------------------------------------------------------------------
// Fused modulated conv via WMMA implicit GEMM.
//   out[b,co,p] = d[b,co]*wscale * sum_{ci,q} W[co,ci,q] * s[b,ci]*x[b,ci,p+q]
//                 + bias[co]   (then optional lrelu*sqrt2, then noise add)
// Workgroup: 256 thr = 8 waves = 4 Cout-subtiles x 2 pixel-pair groups.
// Each wave register-blocks TWO 16-pixel N-subtiles: one A-fragment load feeds
// two v_wmma_f32_16x16x32_f16 issues (2x matrix-op density per byte).
// K is tiled in 32-channel chunks staged to LDS as modulated f16 im2col.
// ---------------------------------------------------------------------------
template <int KSIZE>
__global__ __launch_bounds__(256) void conv_wmma_kernel(
        const float* __restrict__ xin, const float* __restrict__ s,
        const _Float16* __restrict__ w16, const float* __restrict__ dmod,
        const float* __restrict__ bias, const float* __restrict__ nscale,
        float* __restrict__ out,
        int Cin, int Cout, int H, float wscale, int noise_seed, int doAct) {
    constexpr int CHUNK_C = 32;
    constexpr int CHUNK_K = CHUNK_C * KSIZE;   // 288 (3x3) or 32 (1x1): /32 exact
    constexpr int ROWH    = CHUNK_K + 8;       // pad stride (bank spread), 16B mult
    __shared__ __align__(16) _Float16 Bs[64][ROWH];   // 64-pixel tile

    const int b      = blockIdx.z;
    const int HW     = H * H;
    const int p_base = blockIdx.x * 64;
    const int m_base = blockIdx.y * 64;
    const int tid    = threadIdx.x;
    const int lane   = tid & 31;
    const int wave   = tid >> 5;
    const int msub   = wave & 3;               // 4 Cout subtiles (16 rows each)
    const int npair  = wave >> 2;              // 2 pixel-pair groups (2x16 px each)
    const int lhalf  = (lane >= 16) ? 1 : 0;
    const int ncol   = lane & 15;

    v8f acc0, acc1;
    #pragma unroll
    for (int r = 0; r < 8; ++r) { acc0[r] = 0.0f; acc1[r] = 0.0f; }

    int mrow = m_base + msub * 16 + ncol;      // A-frag row for this lane
    if (mrow >= Cout) mrow = 0;                // clamp (store is guarded)
    const _Float16* arow = w16 + (size_t)mrow * ((size_t)Cin * KSIZE);

    const float* sb = s   + (size_t)b * Cin;
    const float* xb = xin + (size_t)b * Cin * HW;

    for (int ci0 = 0; ci0 < Cin; ci0 += CHUNK_C) {
        // ---- stage modulated im2col chunk into LDS (f16): 64px x 32ch ------
        #pragma unroll
        for (int e = 0; e < 8; ++e) {
            int task = tid + e * 256;          // 2048 (pixel,channel) tasks
            int pl   = task & 63;
            int cil  = task >> 6;
            int p    = p_base + pl;
            int ci   = ci0 + cil;
            float sv = sb[ci];
            bool pv  = (p < HW);
            int y = p / H, x = p - y * H;
            const float* xc = xb + (size_t)ci * HW;
            if (KSIZE == 9) {
                #pragma unroll
                for (int q = 0; q < 9; ++q) {
                    int gy = y + q / 3 - 1;
                    int gx = x + q % 3 - 1;
                    float v = 0.0f;
                    if (pv && gy >= 0 && gy < H && gx >= 0 && gx < H)
                        v = xc[gy * H + gx] * sv;
                    Bs[pl][cil * 9 + q] = (_Float16)v;
                }
            } else {
                float v = pv ? xc[y * H + x] * sv : 0.0f;
                Bs[pl][cil] = (_Float16)v;
            }
        }
        __syncthreads();

        // ---- WMMA K-steps over the chunk: 1 A-frag feeds 2 WMMAs ----------
        const int kbase = ci0 * KSIZE;
        #pragma unroll
        for (int ks = 0; ks < CHUNK_K / 32; ++ks) {
            union { v16h v; vf4 f4[2]; } af, bf0, bf1;
            // A-frag (ISA 16-bit A 16x32 layout): two contiguous 16B loads
            int offA = kbase + ks * 32 + lhalf * 8;
            af.f4[0] = *(const vf4*)(arow + offA);
            af.f4[1] = *(const vf4*)(arow + offA + 16);
            // B-frags (ISA 16-bit B 32x16 layout): contiguous 16B LDS loads
            int kk  = ks * 32 + lhalf * 8;
            int pl0 = (npair * 2 + 0) * 16 + ncol;
            int pl1 = (npair * 2 + 1) * 16 + ncol;
            const vf4* bp0 = (const vf4*)(&Bs[pl0][kk]);
            const vf4* bp1 = (const vf4*)(&Bs[pl1][kk]);
            bf0.f4[0] = bp0[0]; bf0.f4[1] = bp0[2];
            bf1.f4[0] = bp1[0]; bf1.f4[1] = bp1[2];
            acc0 = __builtin_amdgcn_wmma_f32_16x16x32_f16(
                    false, af.v, false, bf0.v, (short)0, acc0, false, false);
            acc1 = __builtin_amdgcn_wmma_f32_16x16x32_f16(
                    false, af.v, false, bf1.v, (short)0, acc1, false, false);
        }
        __syncthreads();
    }

    // ---- epilogue: demod, bias, activation, noise --------------------------
    #pragma unroll
    for (int j = 0; j < 2; ++j) {
        int p = p_base + (npair * 2 + j) * 16 + ncol;
        if (p < HW) {
            float g = 0.0f;
            if (nscale) g = gaussian_np((unsigned)noise_seed, (unsigned)b, (unsigned)p);
            #pragma unroll
            for (int r = 0; r < 8; ++r) {
                int co = m_base + msub * 16 + r + lhalf * 8;
                if (co < Cout) {
                    float a = (j == 0) ? acc0[r] : acc1[r];
                    float v = a * dmod[(size_t)b * Cout + co] * wscale + bias[co];
                    if (doAct) v = (v > 0.0f ? v : 0.2f * v) * GAIN;
                    if (nscale) v += g * nscale[co];
                    out[((size_t)b * Cout + co) * HW + p] = v;
                }
            }
        }
    }
}

// ---------------------------------------------------------------------------
// Fused bilinear-up2 (align_corners) + 3x3 [1,2,1]^2/16 blur (zero pad).
// Optional residual add (res may alias dst).
// ---------------------------------------------------------------------------
__global__ void upblur_kernel(const float* __restrict__ x, const float* __restrict__ res,
                              float* __restrict__ dst, int C, int Hin,
                              unsigned long long total) {
    unsigned long long t = (unsigned long long)blockIdx.x * 256ULL + threadIdx.x;
    if (t >= total) return;
    const int Hout = Hin * 2;
    int X = (int)(t % Hout);
    unsigned long long r0 = t / Hout;
    int Y = (int)(r0 % Hout);
    unsigned long long r1 = r0 / Hout;
    int c = (int)(r1 % C);
    int b = (int)(r1 / C);
    const float* xc = x + ((size_t)b * C + c) * Hin * Hin;
    const float ps = (float)(Hin - 1) / (float)(Hout - 1);
    const float kw0 = 0.25f, kw1 = 0.5f;
    float acc = 0.0f;
    #pragma unroll
    for (int dy = 0; dy < 3; ++dy) {
        int Yp = Y + dy - 1;
        if (Yp < 0 || Yp >= Hout) continue;
        float py = (float)Yp * ps;
        int i0 = (int)py;
        int i1 = (i0 + 1 < Hin) ? i0 + 1 : Hin - 1;
        float ty = py - (float)i0;
        float ky = (dy == 1) ? kw1 : kw0;
        #pragma unroll
        for (int dx = 0; dx < 3; ++dx) {
            int Xp = X + dx - 1;
            if (Xp < 0 || Xp >= Hout) continue;
            float px = (float)Xp * ps;
            int j0 = (int)px;
            int j1 = (j0 + 1 < Hin) ? j0 + 1 : Hin - 1;
            float tx = px - (float)j0;
            float kx = (dx == 1) ? kw1 : kw0;
            float v = (1.0f - ty) * ((1.0f - tx) * xc[i0 * Hin + j0] + tx * xc[i0 * Hin + j1])
                    +         ty  * ((1.0f - tx) * xc[i1 * Hin + j0] + tx * xc[i1 * Hin + j1]);
            acc += ky * kx * v;
        }
    }
    if (res) acc += res[t];
    dst[t] = acc;
}

// ---------------------------------------------------------------------------
// Host orchestration
// ---------------------------------------------------------------------------
struct ConvP { const float *aff_b, *aff_w, *b, *w; };

extern "C" void kernel_launch(void* const* d_in, const int* in_sizes, int n_in,
                              void* d_out, int out_size, void* d_ws, size_t ws_size,
                              hipStream_t stream) {
    (void)out_size; (void)ws_size;
    static const int CH[7] = {512, 512, 512, 512, 256, 128, 64};

    // --- locate z (8x512 = 4096 floats is unique among leaves) --------------
    const float* z;
    int idx;
    if (in_sizes[0] == 4096) { z = (const float*)d_in[0]; idx = 1; }
    else                     { z = (const float*)d_in[n_in - 1]; idx = 0; }

    // --- unflatten params: jax pytree order (dict keys sorted) --------------
    // blocks (list) first, then mapping. Leaf dicts sorted alphabetically:
    //   conv_p: aff_b, aff_w, b, w ; block0: const, conv2, noise1, noise2, torgb
    //   blocks1-6: conv1, conv2, noise1, noise2, torgb ; mapping: b, w
    auto F = [&](void) -> const float* { return (const float*)d_in[idx++]; };
    const float* b0_const = F();
    ConvP b0_conv2 = {F(), F(), F(), F()};
    const float* b0_noise1 = F();
    const float* b0_noise2 = F();
    ConvP b0_torgb = {F(), F(), F(), F()};
    struct Blk { ConvP c1, c2, trgb; const float *n1, *n2; } blk[6];
    for (int i = 0; i < 6; ++i) {
        blk[i].c1   = {F(), F(), F(), F()};
        blk[i].c2   = {F(), F(), F(), F()};
        blk[i].n1   = F();
        blk[i].n2   = F();
        blk[i].trgb = {F(), F(), F(), F()};
    }
    const float *mb[8], *mw[8];
    for (int i = 0; i < 8; ++i) { mb[i] = F(); mw[i] = F(); }

    // --- workspace carve ----------------------------------------------------
    size_t off = 0;
    auto carve = [&](size_t bytes) -> void* {
        void* p = (char*)d_ws + off;
        off += (bytes + 255) & ~(size_t)255;
        return p;
    };
    float* arena0 = (float*)carve(268435456ULL);   // 8*128*256^2 f32
    float* arena1 = (float*)carve(268435456ULL);
    float* rgbA   = (float*)carve(6291456ULL);     // 8*3*256^2 f32
    float* rgbB   = (float*)carve(6291456ULL);
    float* mapA   = (float*)carve(16384);
    float* mapB   = (float*)carve(16384);
    float* sbuf   = (float*)carve(16384);
    float* dbuf   = (float*)carve(16384);
    float* wsq    = (float*)carve(1048576ULL);     // 512*512 f32
    _Float16* w16 = (_Float16*)carve(4718848ULL);  // 512*512*9 f16 (+pad)

    // --- mapping network ----------------------------------------------------
    pixnorm_kernel<<<8, 256, 0, stream>>>(z, mapA);
    float* cur = mapA; float* nxt = mapB;
    for (int l = 0; l < 8; ++l) {
        eqlin_kernel<<<(8 * 512 + 255) / 256, 256, 0, stream>>>(
            cur, mw[l], mb[l], nxt, 512, 512, INV_SQRT512, 1);
        float* t = cur; cur = nxt; nxt = t;
    }
    const float* wlat = cur;

    // --- modulated conv helper ---------------------------------------------
    auto modconv = [&](const ConvP& cp, int cin, int cout, int k2,
                       const float* xin, float* xout, int H,
                       int doAct, const float* nsc, int nidx) {
        eqlin_kernel<<<(8 * cin + 255) / 256, 256, 0, stream>>>(
            wlat, cp.aff_w, cp.aff_b, sbuf, cin, 512, INV_SQRT512, 0);
        wprep_kernel<<<(cout * cin + 255) / 256, 256, 0, stream>>>(
            cp.w, w16, wsq, cout, cin, k2);
        float wscale = 1.0f / sqrtf((float)(cin * k2));
        demod_kernel<<<(8 * cout + 255) / 256, 256, 0, stream>>>(
            sbuf, wsq, dbuf, cout, cin, wscale);
        dim3 g((H * H + 63) / 64, (cout + 63) / 64, 8);
        if (k2 == 9)
            conv_wmma_kernel<9><<<g, 256, 0, stream>>>(
                xin, sbuf, w16, dbuf, cp.b, nsc, xout, cin, cout, H, wscale, nidx, doAct);
        else
            conv_wmma_kernel<1><<<g, 256, 0, stream>>>(
                xin, sbuf, w16, dbuf, cp.b, nsc, xout, cin, cout, H, wscale, nidx, doAct);
    };

    // --- block 0 (4x4) ------------------------------------------------------
    const_noise_kernel<<<(8 * 512 * 16 + 255) / 256, 256, 0, stream>>>(
        b0_const, b0_noise1, arena0);
    modconv(b0_conv2, 512, 512, 9, arena0, arena1, 4, 1, b0_noise2, 1);
    float* h = arena1; float* other = arena0;
    modconv(b0_torgb, 512, 3, 1, h, rgbA, 4, 0, nullptr, 0);
    float* rgb = rgbA; float* rgbO = rgbB;

    // --- blocks 1..6 --------------------------------------------------------
    int nidx = 2;
    for (int i = 1; i < 7; ++i) {
        int cin = CH[i - 1], cout = CH[i];
        int Hin = 4 << (i - 1), Hout = Hin * 2;
        // U = blur(up2(h)) -> other (fused)
        unsigned long long tot = (unsigned long long)8 * cin * Hout * Hout;
        upblur_kernel<<<(unsigned)((tot + 255) / 256), 256, 0, stream>>>(
            h, nullptr, other, cin, Hin, tot);
        modconv(blk[i - 1].c1, cin, cout, 9, other, h, Hout, 1, blk[i - 1].n1, nidx);
        modconv(blk[i - 1].c2, cout, cout, 9, h, other, Hout, 1, blk[i - 1].n2, nidx + 1);
        nidx += 2;
        { float* t = h; h = other; other = t; }   // h = conv2 output
        // rgb skip path
        modconv(blk[i - 1].trgb, cout, 3, 1, h, rgbO, Hout, 0, nullptr, 0);
        float* dst = (i == 6) ? (float*)d_out : rgbO;   // in-place add OK
        unsigned long long rtot = (unsigned long long)8 * 3 * Hout * Hout;
        upblur_kernel<<<(unsigned)((rtot + 255) / 256), 256, 0, stream>>>(
            rgb, rgbO, dst, 3, Hin, rtot);
        float* t = rgb; rgb = rgbO; rgbO = t;
    }
}
